// OptimizedTransformerBlock_19662360281643
// MI455X (gfx1250) — matrix operand order repaired and checked by reading, not compiled, not run on hardware
//
#include <hip/hip_runtime.h>

typedef __bf16 bf16;
typedef __attribute__((ext_vector_type(16))) __bf16 bf16x16;
typedef __attribute__((ext_vector_type(8)))  float  f32x8;

union FragU {
  bf16x16 v;
  struct { uint4 lo, hi; } q;
};

__device__ __forceinline__ bf16 f2bf(float f) {
  unsigned u = __builtin_bit_cast(unsigned, f);
  u += 0x7FFFu + ((u >> 16) & 1u);           // round-to-nearest-even
  unsigned short h = (unsigned short)(u >> 16);
  return __builtin_bit_cast(bf16, h);
}

__device__ __forceinline__ f32x8 wmma_bf16(const FragU& a, const FragU& b, f32x8 c) {
  // D = A(16x32 bf16) * B(32x16 bf16) + C(16x16 f32)
  return __builtin_amdgcn_wmma_f32_16x16x32_bf16(false, a.v, false, b.v,
                                                 (short)0, c, false, false);
}

// CDNA5 async cache->LDS copy (bypasses VGPRs, tracked by ASYNCcnt).
// LDS aperture: generic-pointer low 32 bits are the LDS byte offset.
__device__ __forceinline__ void async_copy_b128(const bf16* g, bf16* l) {
  unsigned loff = (unsigned)(unsigned long long)l;
  asm volatile("global_load_async_to_lds_b128 %0, %1, off"
               :: "v"(loff), "v"((unsigned long long)g) : "memory");
}
__device__ __forceinline__ void wait_async0() {
  asm volatile("s_wait_asynccnt 0" ::: "memory");
}

// ---------------------------------------------------------------------------
// Generic bf16 GEMM: C[M][N] = A[M][K] @ Bt[N][K]^T + bias, optional ReLU.
// Block tile 128x128, 8 waves of 32 (wave tile 64x32), K step 32.
// Double-buffered LDS fed by async global->LDS DMA; one barrier per K-step.
// ---------------------------------------------------------------------------
template<bool RELU, bool OUT_BF16>
__global__ __launch_bounds__(256) void gemm_bf16_kernel(
    const bf16* __restrict__ A, const bf16* __restrict__ Bt,
    const float* __restrict__ bias, void* __restrict__ Cout,
    int M, int N, int K)
{
  constexpr int LDSTR = 40;                      // 32 + 8 pad (80B rows, conflict-free b128 reads)
  __shared__ bf16 As[2][128 * LDSTR];
  __shared__ bf16 Bs[2][128 * LDSTR];

  const int tid  = threadIdx.x;
  const int lane = tid & 31;
  const int wave = tid >> 5;
  const int wm   = wave >> 2;                    // 0..1 : 64-row stripe
  const int wn   = wave & 3;                     // 0..3 : 32-col stripe
  const long bm  = (long)blockIdx.y * 128;
  const long bn  = (long)blockIdx.x * 128;

  // Each thread owns 2 x 16B chunks of the A tile and 2 of the B tile.
  int cr[2], co[2];
  #pragma unroll
  for (int i = 0; i < 2; ++i) {
    int c = tid + i * 256;                       // 512 chunks per 128x32 tile
    cr[i] = c >> 2;                              // row in tile
    co[i] = (c & 3) * 8;                         // element offset within 32-elem row
  }

  auto issue = [&](int buf, int kk) {
    #pragma unroll
    for (int i = 0; i < 2; ++i) {
      async_copy_b128(A  + (bm + cr[i]) * (long)K + kk + co[i],
                      &As[buf][cr[i] * LDSTR + co[i]]);
      async_copy_b128(Bt + (bn + cr[i]) * (long)K + kk + co[i],
                      &Bs[buf][cr[i] * LDSTR + co[i]]);
    }
  };

  issue(0, 0);

  f32x8 acc[4][2] = {};
  const int nsteps = K >> 5;

  for (int s = 0; s < nsteps; ++s) {
    const int cur = s & 1;
    wait_async0();                               // own async writes to LDS done
    __syncthreads();                             // everyone's writes visible; prev reads done
    if (s + 1 < nsteps) issue(cur ^ 1, (s + 1) << 5);

    FragU bfr[2];
    #pragma unroll
    for (int nt = 0; nt < 2; ++nt) {
      // B operand: lane holds column n = 16*nt + lane%16, K {0..7,16..23} / {8..15,24..31}
      const bf16* p = &Bs[cur][(wn * 32 + nt * 16 + (lane & 15)) * LDSTR + ((lane < 16) ? 0 : 8)];
      bfr[nt].q.lo = *(const uint4*)p;
      bfr[nt].q.hi = *(const uint4*)(p + 16);
    }
    #pragma unroll
    for (int mt = 0; mt < 4; ++mt) {
      // A operand: lane holds row m = 16*mt + lane%16, K {0..7,16..23} / {8..15,24..31}
      FragU af;
      const bf16* p = &As[cur][(wm * 64 + mt * 16 + (lane & 15)) * LDSTR + ((lane < 16) ? 0 : 8)];
      af.q.lo = *(const uint4*)p;
      af.q.hi = *(const uint4*)(p + 16);
      #pragma unroll
      for (int nt = 0; nt < 2; ++nt)
        acc[mt][nt] = wmma_bf16(af, bfr[nt], acc[mt][nt]);
    }
  }

  // Epilogue: C layout — VGPR r holds (M = r + 8*(lane>=16), N = lane%16)
  const int rowoff = (lane >> 4) << 3;
  #pragma unroll
  for (int mt = 0; mt < 4; ++mt) {
    #pragma unroll
    for (int nt = 0; nt < 2; ++nt) {
      long col = bn + wn * 32 + nt * 16 + (lane & 15);
      float bv = bias[col];
      long rowb = bm + wm * 64 + mt * 16 + rowoff;
      #pragma unroll
      for (int r = 0; r < 8; ++r) {
        float v = acc[mt][nt][r] + bv;
        if (RELU) v = fmaxf(v, 0.f);
        if (OUT_BF16) ((bf16*)Cout)[(rowb + r) * (long)N + col] = f2bf(v);
        else          ((float*)Cout)[(rowb + r) * (long)N + col] = v;
      }
    }
  }
}

// ---------------------------------------------------------------------------
// Block-local window attention: one wave per (batch, head, window).
// W=32 tokens, dh=64. scores = Q Kt / 8 -> softmax -> P V, all via WMMA.
// ---------------------------------------------------------------------------
__global__ __launch_bounds__(256) void attn_kernel(
    const bf16* __restrict__ Q, const bf16* __restrict__ Kb,
    const bf16* __restrict__ V, bf16* __restrict__ O)
{
  constexpr int S = 8192, D = 512, DH = 64;
  constexpr int PST = 40;                         // padded LDS row stride (elems)
  __shared__ bf16 Pbuf[8 * 32 * PST];             // per-wave 32x32 P tile
  __shared__ bf16 Vtb [8 * 64 * PST];             // per-wave 64x32 V^T tile

  const int tid  = threadIdx.x;
  const int lane = tid & 31;
  const int wave = tid >> 5;
  bf16* Ps = &Pbuf[wave * 32 * PST];
  bf16* Vs = &Vtb [wave * 64 * PST];

  const int gw = blockIdx.x * 8 + wave;           // 0..16383
  const int b  = gw >> 11;                        // H*nW = 2048
  const int h  = (gw >> 8) & 7;
  const int w  = gw & 255;

  const long rowbase = (long)b * S + (long)w * 32;
  const bf16* Qp = Q  + rowbase * D + h * DH;
  const bf16* Kp = Kb + rowbase * D + h * DH;
  const bf16* Vp = V  + rowbase * D + h * DH;

  // --- load Q (A operand) and K (B operand) fragments straight from global ---
  FragU aq[2][2], bk[2][2];
  #pragma unroll
  for (int mt = 0; mt < 2; ++mt)
    #pragma unroll
    for (int ks = 0; ks < 2; ++ks) {
      const bf16* p = Qp + (long)(mt * 16 + (lane & 15)) * D + ks * 32 + ((lane < 16) ? 0 : 8);
      aq[mt][ks].q.lo = *(const uint4*)p;
      aq[mt][ks].q.hi = *(const uint4*)(p + 16);
    }
  #pragma unroll
  for (int nt = 0; nt < 2; ++nt)
    #pragma unroll
    for (int ks = 0; ks < 2; ++ks) {
      const bf16* p = Kp + (long)(nt * 16 + (lane & 15)) * D + ks * 32 + ((lane < 16) ? 0 : 8);
      bk[nt][ks].q.lo = *(const uint4*)p;
      bk[nt][ks].q.hi = *(const uint4*)(p + 16);
    }

  // --- scores = Q K^T (32x32, K-dim = dh = 64) ---
  f32x8 sc[2][2] = {};
  #pragma unroll
  for (int ks = 0; ks < 2; ++ks)
    #pragma unroll
    for (int mt = 0; mt < 2; ++mt)
      #pragma unroll
      for (int nt = 0; nt < 2; ++nt)
        sc[mt][nt] = wmma_bf16(aq[mt][ks], bk[nt][ks], sc[mt][nt]);

  // --- softmax over 32 cols; row elements live in one 16-lane half ---
  constexpr float scale = 0.125f;                 // 1/sqrt(64)
  #pragma unroll
  for (int mt = 0; mt < 2; ++mt)
    #pragma unroll
    for (int r = 0; r < 8; ++r) {
      float a0 = sc[mt][0][r] * scale;
      float a1 = sc[mt][1][r] * scale;
      float m = fmaxf(a0, a1);
      #pragma unroll
      for (int o = 1; o < 16; o <<= 1) m = fmaxf(m, __shfl_xor(m, o, 32));
      float e0 = __expf(a0 - m), e1 = __expf(a1 - m);
      float sm = e0 + e1;
      #pragma unroll
      for (int o = 1; o < 16; o <<= 1) sm += __shfl_xor(sm, o, 32);
      float inv = 1.0f / sm;
      sc[mt][0][r] = e0 * inv;
      sc[mt][1][r] = e1 * inv;
    }

  // --- stage P (C layout -> row-major LDS) and V^T ( [dh][tok] ) ---
  const int rowoff = (lane >> 4) << 3;
  #pragma unroll
  for (int mt = 0; mt < 2; ++mt)
    #pragma unroll
    for (int nt = 0; nt < 2; ++nt)
      #pragma unroll
      for (int r = 0; r < 8; ++r)
        Ps[(mt * 16 + r + rowoff) * PST + nt * 16 + (lane & 15)] = f2bf(sc[mt][nt][r]);

  #pragma unroll
  for (int dc = 0; dc < DH; dc += 8) {
    union { uint4 q; bf16 e[8]; } u;
    u.q = *(const uint4*)(Vp + (long)lane * D + dc);
    #pragma unroll
    for (int j = 0; j < 8; ++j) Vs[(dc + j) * PST + lane] = u.e[j];
  }
  __syncthreads();

  // --- O = P(32x32) @ V(32x64): single K-step of 32 ---
  FragU ap[2], bv[4];
  #pragma unroll
  for (int mt = 0; mt < 2; ++mt) {
    const bf16* p = Ps + (mt * 16 + (lane & 15)) * PST + ((lane < 16) ? 0 : 8);
    ap[mt].q.lo = *(const uint4*)p;
    ap[mt].q.hi = *(const uint4*)(p + 16);
  }
  #pragma unroll
  for (int nt = 0; nt < 4; ++nt) {
    const bf16* p = Vs + (nt * 16 + (lane & 15)) * PST + ((lane < 16) ? 0 : 8);
    bv[nt].q.lo = *(const uint4*)p;
    bv[nt].q.hi = *(const uint4*)(p + 16);
  }
  f32x8 oa[2][4] = {};
  #pragma unroll
  for (int mt = 0; mt < 2; ++mt)
    #pragma unroll
    for (int nt = 0; nt < 4; ++nt)
      oa[mt][nt] = wmma_bf16(ap[mt], bv[nt], oa[mt][nt]);

  bf16* Op = O + rowbase * D + h * DH;
  #pragma unroll
  for (int mt = 0; mt < 2; ++mt)
    #pragma unroll
    for (int nt = 0; nt < 4; ++nt)
      #pragma unroll
      for (int r = 0; r < 8; ++r)
        Op[(long)(mt * 16 + r + rowoff) * D + nt * 16 + (lane & 15)] = f2bf(oa[mt][nt][r]);
}

// ---------------------------------------------------------------------------
// Residual add + LayerNorm over D=512. One row per 128-thread block.
// Emits f32 (for next residual) and optional bf16 (for next GEMM).
// ---------------------------------------------------------------------------
__global__ __launch_bounds__(128) void add_ln_kernel(
    const float* __restrict__ X, const float* __restrict__ Y,
    const float* __restrict__ G, const float* __restrict__ Bv,
    float* __restrict__ Of, bf16* __restrict__ Ob)
{
  const int tid = threadIdx.x;
  const long base = (long)blockIdx.x * 512;
  float v[4];
  float s = 0.f, ss = 0.f;
  #pragma unroll
  for (int i = 0; i < 4; ++i) {
    int c = tid + i * 128;
    float t = X[base + c] + Y[base + c];
    v[i] = t; s += t; ss += t * t;
  }
  #pragma unroll
  for (int o = 16; o > 0; o >>= 1) { s += __shfl_xor(s, o, 32); ss += __shfl_xor(ss, o, 32); }
  __shared__ float sh[8];
  if ((tid & 31) == 0) { sh[tid >> 5] = s; sh[4 + (tid >> 5)] = ss; }
  __syncthreads();
  s  = sh[0] + sh[1] + sh[2] + sh[3];
  ss = sh[4] + sh[5] + sh[6] + sh[7];
  float mean = s * (1.f / 512.f);
  float var  = ss * (1.f / 512.f) - mean * mean;
  float r = rsqrtf(var + 1e-5f);
  #pragma unroll
  for (int i = 0; i < 4; ++i) {
    int c = tid + i * 128;
    float o = (v[i] - mean) * r * G[c] + Bv[c];
    Of[base + c] = o;
    if (Ob) Ob[base + c] = f2bf(o);
  }
}

// ---------------------------------------------------------------------------
// f32 -> bf16 elementwise, and transposing weight convert W[K][N] -> Wt[N][K].
// ---------------------------------------------------------------------------
__global__ void cvt_bf16_kernel(const float* __restrict__ in, bf16* __restrict__ out, int n) {
  int i = blockIdx.x * blockDim.x + threadIdx.x;
  int stride = gridDim.x * blockDim.x;
  for (; i < n; i += stride) out[i] = f2bf(in[i]);
}

__global__ void cvt_transpose_kernel(const float* __restrict__ W, bf16* __restrict__ Wt,
                                     int Kdim, int Ndim) {
  int i = blockIdx.x * blockDim.x + threadIdx.x;
  if (i >= Kdim * Ndim) return;
  int n = i / Kdim, k = i - n * Kdim;
  Wt[i] = f2bf(W[(long)k * Ndim + n]);
}

// ---------------------------------------------------------------------------
extern "C" void kernel_launch(void* const* d_in, const int* in_sizes, int n_in,
                              void* d_out, int out_size, void* d_ws, size_t ws_size,
                              hipStream_t stream) {
  (void)in_sizes; (void)n_in; (void)out_size; (void)ws_size;
  const float* x   = (const float*)d_in[0];
  const float* Wq  = (const float*)d_in[1];
  const float* bq  = (const float*)d_in[2];
  const float* Wk  = (const float*)d_in[3];
  const float* bk  = (const float*)d_in[4];
  const float* Wv  = (const float*)d_in[5];
  const float* bv  = (const float*)d_in[6];
  const float* Wo  = (const float*)d_in[7];
  const float* bo  = (const float*)d_in[8];
  const float* g1  = (const float*)d_in[9];
  const float* be1 = (const float*)d_in[10];
  const float* W1  = (const float*)d_in[11];
  const float* b1  = (const float*)d_in[12];
  const float* W2  = (const float*)d_in[13];
  const float* b2  = (const float*)d_in[14];
  const float* g2  = (const float*)d_in[15];
  const float* be2 = (const float*)d_in[16];

  constexpr long M = 65536;  // B*S
  constexpr long D = 512, F = 2048;

  char* ws = (char*)d_ws;
  size_t off = 0;
  bf16* xbf  = (bf16*)(ws + off); off += (size_t)M * D * 2;     // 64 MB
  bf16* Qb   = (bf16*)(ws + off); off += (size_t)M * D * 2;     // Q (reused as hidden)
  bf16* Kbb  = (bf16*)(ws + off); off += (size_t)M * D * 2;
  bf16* Vb   = (bf16*)(ws + off); off += (size_t)M * D * 2;
  bf16* AO   = (bf16*)(ws + off); off += (size_t)M * D * 2;
  bf16* hidden = Qb;                                            // 4*M*D*2 == M*F*2
  float* proj = (float*)(ws + off); off += (size_t)M * D * 4;   // attn proj / ffn2 out
  float* x1f  = (float*)(ws + off); off += (size_t)M * D * 4;
  bf16*  x1b  = (bf16*)(ws + off); off += (size_t)M * D * 2;
  bf16* wqT  = (bf16*)(ws + off); off += (size_t)D * D * 2;
  bf16* wkT  = (bf16*)(ws + off); off += (size_t)D * D * 2;
  bf16* wvT  = (bf16*)(ws + off); off += (size_t)D * D * 2;
  bf16* woT  = (bf16*)(ws + off); off += (size_t)D * D * 2;
  bf16* w1T  = (bf16*)(ws + off); off += (size_t)F * D * 2;
  bf16* w2T  = (bf16*)(ws + off); off += (size_t)D * F * 2;

  // --- converts ---
  cvt_bf16_kernel<<<4096, 256, 0, stream>>>(x, xbf, (int)(M * D));
  cvt_transpose_kernel<<<(D * D + 255) / 256, 256, 0, stream>>>(Wq, wqT, (int)D, (int)D);
  cvt_transpose_kernel<<<(D * D + 255) / 256, 256, 0, stream>>>(Wk, wkT, (int)D, (int)D);
  cvt_transpose_kernel<<<(D * D + 255) / 256, 256, 0, stream>>>(Wv, wvT, (int)D, (int)D);
  cvt_transpose_kernel<<<(D * D + 255) / 256, 256, 0, stream>>>(Wo, woT, (int)D, (int)D);
  cvt_transpose_kernel<<<(D * F + 255) / 256, 256, 0, stream>>>(W1, w1T, (int)D, (int)F);
  cvt_transpose_kernel<<<(D * F + 255) / 256, 256, 0, stream>>>(W2, w2T, (int)F, (int)D);

  dim3 blk(256);
  // --- QKV projections ---
  gemm_bf16_kernel<false, true><<<dim3(4, 512), blk, 0, stream>>>(xbf, wqT, bq, Qb,  (int)M, (int)D, (int)D);
  gemm_bf16_kernel<false, true><<<dim3(4, 512), blk, 0, stream>>>(xbf, wkT, bk, Kbb, (int)M, (int)D, (int)D);
  gemm_bf16_kernel<false, true><<<dim3(4, 512), blk, 0, stream>>>(xbf, wvT, bv, Vb,  (int)M, (int)D, (int)D);
  // --- windowed attention ---
  attn_kernel<<<2048, 256, 0, stream>>>(Qb, Kbb, Vb, AO);
  // --- output projection (f32 out) ---
  gemm_bf16_kernel<false, false><<<dim3(4, 512), blk, 0, stream>>>(AO, woT, bo, proj, (int)M, (int)D, (int)D);
  // --- residual + LN1 ---
  add_ln_kernel<<<65536, 128, 0, stream>>>(x, proj, g1, be1, x1f, x1b);
  // --- FFN ---
  gemm_bf16_kernel<true, true ><<<dim3(16, 512), blk, 0, stream>>>(x1b, w1T, b1, hidden, (int)M, (int)F, (int)D);
  gemm_bf16_kernel<false, false><<<dim3(4, 512), blk, 0, stream>>>(hidden, w2T, b2, proj, (int)M, (int)D, (int)F);
  // --- residual + LN2 -> d_out (f32) ---
  add_ln_kernel<<<65536, 128, 0, stream>>>(x1f, proj, g2, be2, (float*)d_out, (bf16*)nullptr);
}
